// SOABlock_39290360824061
// MI455X (gfx1250) — matrix operand order repaired and checked by reading, not compiled, not run on hardware
//
#include <hip/hip_runtime.h>
#include <cstdint>
#include <cstddef>

// ---------------- problem constants ----------------
#define B_    16
#define C_    512
#define MID_  256
#define HW_   2304            // 48*48
#define NT_   144             // HW_/16  (16-row n-tiles)
#define MCH_  72              // HW_/32  (32-col m-chunks)
#define EPS_  1e-5f
#define SCALE_ 0.0625f        // mid^-0.5 = 1/16

typedef __attribute__((ext_vector_type(16))) __bf16 bf16x16;
typedef __attribute__((ext_vector_type(8)))  __bf16 bf16x8;
typedef __attribute__((ext_vector_type(8)))  float  f32x8;
typedef __attribute__((ext_vector_type(4)))  float  f32x4;
typedef __attribute__((ext_vector_type(4)))  uint32_t u32x4;
typedef __attribute__((ext_vector_type(8)))  int      i32x8;
typedef __attribute__((ext_vector_type(4)))  int      i32x4;

#define CAT16(lo,hi) __builtin_shufflevector((lo),(hi),0,1,2,3,4,5,6,7,8,9,10,11,12,13,14,15)

__device__ __forceinline__ f32x8 wmma_bf16(bf16x16 a, bf16x16 b, f32x8 c) {
  // D = A(16x32 bf16) * B(32x16 bf16) + C(16x16 f32)
  return __builtin_amdgcn_wmma_f32_16x16x32_bf16(false, a, false, b, (short)0, c, false, false);
}

// D-layout rows live per 16-lane half (wave32): xor masks 1,2,4,8 reduce a row
__device__ __forceinline__ float rowmax16(float v) {
  v = fmaxf(v, __shfl_xor(v, 1, 32));
  v = fmaxf(v, __shfl_xor(v, 2, 32));
  v = fmaxf(v, __shfl_xor(v, 4, 32));
  v = fmaxf(v, __shfl_xor(v, 8, 32));
  return v;
}
__device__ __forceinline__ float rowsum16(float v) {
  v += __shfl_xor(v, 1, 32);
  v += __shfl_xor(v, 2, 32);
  v += __shfl_xor(v, 4, 32);
  v += __shfl_xor(v, 8, 32);
  return v;
}

// CDNA5 async global->LDS DMA (ASYNCcnt-tracked).  vdst = LDS byte address,
// vaddr = 64-bit global address ("GV" mode, saddr = off/NULL).
__device__ __forceinline__ void g2l_async_b128(uint32_t lds_off, const void* gsrc) {
  asm volatile("global_load_async_to_lds_b128 %0, %1, off"
               :: "v"(lds_off), "v"(gsrc) : "memory");
}

// CDNA5 Tensor Data Mover: one-instruction 2D tile DMA global->LDS.
// Descriptor packing per ISA 8.3/8.4 (group0: count/lds/global/type,
// group1: data_size + tensor/tile dims + dim0 stride).  data_size = 2 B.
// tile_d1 == 0 -> 1-D tile.  Groups 2/3 zero (tensor <= 2D).
// This toolchain's builtin takes 6 args: (g0, g1, g2, g3, g_extra, cpol).
__device__ __forceinline__ void tdm_load_2d(uint32_t lds_addr, const void* gaddr,
                                            uint32_t tile_d0, uint32_t tile_d1,
                                            uint64_t stride_d0_elems) {
  uint64_t ga = (uint64_t)(uintptr_t)gaddr;
  u32x4 g0;
  g0[0] = 1u;                                        // count=1, user descriptor
  g0[1] = lds_addr;                                  // LDS byte address
  g0[2] = (uint32_t)(ga & 0xffffffffu);              // global addr [31:0]
  g0[3] = (uint32_t)((ga >> 32) & 0x01ffffffu) | (2u << 30);  // [56:32] | type=2
  const uint32_t TD0 = 1u << 24;                     // huge tensor dims: no OOB
  const uint32_t TD1 = 1u << 24;
  i32x8 g1;
  g1[0] = (int)(1u << 16);                           // data_size = 1 (2 bytes)
  g1[1] = (int)((TD0 & 0xffffu) << 16);              // tensor_dim0 [15:0]
  g1[2] = (int)((TD0 >> 16) | ((TD1 & 0xffffu) << 16));
  g1[3] = (int)((TD1 >> 16) | (tile_d0 << 16));      // tile_dim0
  g1[4] = (int)(tile_d1 & 0xffffu);                  // tile_dim1 (0 = unused)
  g1[5] = (int)(uint32_t)stride_d0_elems;            // tensor_dim0_stride[31:0]
  g1[6] = (int)((uint32_t)(stride_d0_elems >> 32) & 0xffffu);
  g1[7] = 0;
  i32x4 z4 = {0, 0, 0, 0};
  i32x8 z8i = {0, 0, 0, 0, 0, 0, 0, 0};
  __builtin_amdgcn_tensor_load_to_lds(g0, g1, z4, z4, z8i, 0);
}

// =====================================================================
// Kernel 0: fold BN into f/g weights; pre-swizzle all weights into WMMA
// fragment layouts (A-frag for f/g/h, B-frag for v), fold biases.
//   A-frag (16x32 bf16): lane l, elem e: M = l%16,
//       K = (l/16)*16 + e
//   B-frag (32x16 bf16): lane l, elem e: N = l%16, K = (l/16)*16 + e
// =====================================================================
__global__ __launch_bounds__(256)
void prep_kernel(const float* __restrict__ f_w, const float* __restrict__ f_b,
                 const float* __restrict__ f_gamma, const float* __restrict__ f_beta,
                 const float* __restrict__ f_mean, const float* __restrict__ f_var,
                 const float* __restrict__ g_w, const float* __restrict__ g_b,
                 const float* __restrict__ g_gamma, const float* __restrict__ g_beta,
                 const float* __restrict__ g_mean, const float* __restrict__ g_var,
                 const float* __restrict__ h_w, const float* __restrict__ v_w,
                 __bf16* __restrict__ wfF, __bf16* __restrict__ wfG,
                 __bf16* __restrict__ wfH, __bf16* __restrict__ wfV,
                 float* __restrict__ biasF, float* __restrict__ biasG) {
  int id = blockIdx.x * 256 + threadIdx.x;
  if (id < 393216) {                       // 3 mats * (16ct * 16kc * 32lane * 16e)
    int mat = id >> 17;
    int rem = id & 131071;
    int e    =  rem        & 15;
    int lane = (rem >> 4)  & 31;
    int kc   = (rem >> 9)  & 15;
    int ct   =  rem >> 13;
    int oc  = ct * 16 + (lane & 15);
    int off = (lane >> 4) * 8;
    int k   = kc * 32 + off * 2 + e;
    const float* w = (mat == 0) ? f_w : (mat == 1) ? g_w : h_w;
    float scale = 1.0f;
    if (mat == 0)      scale = f_gamma[oc] * rsqrtf(f_var[oc] + EPS_);
    else if (mat == 1) scale = g_gamma[oc] * rsqrtf(g_var[oc] + EPS_);
    __bf16 v = (__bf16)(w[oc * C_ + k] * scale);
    ((mat == 0) ? wfF : (mat == 1) ? wfG : wfH)[rem] = v;
  } else if (id < 524288) {                // V: 32ot * 8kc * 32lane * 16e (B-frag)
    int rem  = id - 393216;
    int e    =  rem        & 15;
    int lane = (rem >> 4)  & 31;
    int kc   = (rem >> 9)  & 7;
    int ot   =  rem >> 12;
    int oc = ot * 16 + (lane & 15);
    int k  = kc * 32 + (lane >> 4) * 16 + e;
    wfV[rem] = (__bf16)v_w[oc * MID_ + k];
  } else if (id < 524800) {                // folded biases
    int oc = id - 524288;
    if (oc < 256) {
      float s = f_gamma[oc] * rsqrtf(f_var[oc] + EPS_);
      biasF[oc] = (f_b[oc] - f_mean[oc]) * s + f_beta[oc];
    } else {
      int o = oc - 256;
      float s = g_gamma[o] * rsqrtf(g_var[o] + EPS_);
      biasG[o] = (g_b[o] - g_mean[o]) * s + g_beta[o];
    }
  }
}

// =====================================================================
// Kernel 1: f/g/h projections.  Grid (144, 16), 128 thr (4 waves).
// Wave w computes c-tiles [4w, 4w+4) for one (b, n-tile); the x B-frag
// is shared by all 12 WMMAs per k-chunk.
// Outputs: fT/gT as [b][n][k] bf16 (post BN+ReLU), h as [b][c][m] bf16.
// =====================================================================
__global__ __launch_bounds__(128)
void proj_kernel(const float* __restrict__ x,
                 const __bf16* __restrict__ wfF, const __bf16* __restrict__ wfG,
                 const __bf16* __restrict__ wfH,
                 const float* __restrict__ biasF, const float* __restrict__ biasG,
                 const float* __restrict__ h_b,
                 __bf16* __restrict__ fT, __bf16* __restrict__ gT,
                 __bf16* __restrict__ hM) {
  const int nt = blockIdx.x, b = blockIdx.y;
  const int lane = threadIdx.x & 31, wv = threadIdx.x >> 5;
  const int half = lane >> 4, l16 = lane & 15;
  const int ct0 = wv * 4;

  f32x8 aF[4], aG[4], aH[4];
  const f32x8 z8 = {0.f,0.f,0.f,0.f,0.f,0.f,0.f,0.f};
#pragma unroll
  for (int t = 0; t < 4; ++t) { aF[t] = z8; aG[t] = z8; aH[t] = z8; }

  // x B-frag: lane = col n0+l16, k = kc*32 + half*16 + j (strided in C-major x)
  const float* xp = x + ((size_t)b * C_ + half * 16) * HW_ + (size_t)nt * 16 + l16;

  for (int kc = 0; kc < 16; ++kc) {
    bf16x16 bx;
    const float* xq = xp + (size_t)kc * 32 * HW_;
#pragma unroll
    for (int j = 0; j < 16; ++j) bx[j] = (__bf16)xq[(size_t)j * HW_];
#pragma unroll
    for (int t = 0; t < 4; ++t) {
      size_t fo = (((size_t)(ct0 + t) * 16 + kc) * 32 + lane) * 16;
      aF[t] = wmma_bf16(*(const bf16x16*)(wfF + fo), bx, aF[t]);
      aG[t] = wmma_bf16(*(const bf16x16*)(wfG + fo), bx, aG[t]);
      aH[t] = wmma_bf16(*(const bf16x16*)(wfH + fo), bx, aH[t]);
    }
  }

#pragma unroll
  for (int t = 0; t < 4; ++t) {
    int ct = ct0 + t;
    int ocb = ct * 16 + 8 * half;      // D row (oc) = ct*16 + r + 8*half
    bf16x8 pf, pg;
#pragma unroll
    for (int r = 0; r < 8; ++r) {
      int oc = ocb + r;
      pf[r] = (__bf16)fmaxf(aF[t][r] + biasF[oc], 0.f);
      pg[r] = (__bf16)fmaxf(aG[t][r] + biasG[oc], 0.f);
    }
    size_t rowo = ((size_t)b * HW_ + (size_t)nt * 16 + l16) * MID_ + ocb;
    *(bf16x8*)(fT + rowo) = pf;       // 8 consecutive k -> one 16B store
    *(bf16x8*)(gT + rowo) = pg;
#pragma unroll
    for (int r = 0; r < 8; ++r) {
      int oc = ocb + r;
      hM[((size_t)b * MID_ + oc) * HW_ + (size_t)nt * 16 + l16] =
          (__bf16)(aH[t][r] + h_b[oc]);
    }
  }
}

// =====================================================================
// Kernel 2: flash attention.  Grid (36, 16), 128 thr = 4 waves.
// Wave w owns n-tile (4*blockIdx.x + w): full online softmax, no combine.
// G/H m-chunks (32 KB) are staged by TWO TDM descriptors issued by wave 0
// (TENSORcnt double-buffered); F tiles async-staged once (ASYNCcnt).
// After the loop the double buffer is reused as f32 scratch to emit
// z[b][n][k] bf16 coalesced.
// =====================================================================
__global__ __launch_bounds__(128)
void attn_kernel(const __bf16* __restrict__ fT, const __bf16* __restrict__ gT,
                 const __bf16* __restrict__ hM, __bf16* __restrict__ zT) {
  const int nt4 = blockIdx.x, b = blockIdx.y;
  const int tid = threadIdx.x, lane = tid & 31, wv = tid >> 5;
  const int half = lane >> 4, l16 = lane & 15;

  __shared__ __attribute__((aligned(32))) char   Sbuf[2][32768]; // G(16K)+H(16K) x2
  __shared__ __attribute__((aligned(32))) __bf16 Flds[64 * 256]; // 32 KB
  __shared__ __attribute__((aligned(32))) __bf16 Pb[4][16 * 32]; // 4 KB

  const uint32_t sb0 = (uint32_t)(uintptr_t)&Sbuf[0][0];
  const uint32_t fds = (uint32_t)(uintptr_t)&Flds[0];

  const __bf16* gbase = gT + (size_t)b * HW_ * MID_;   // [m][k] rows
  const __bf16* hbase = hM + (size_t)b * MID_ * HW_;   // [c][m] rows

  // ---- prologue: async-stage F (64 contiguous n-rows); TDM chunk 0 ----
  {
    const __bf16* fsrc = fT + ((size_t)b * HW_ + (size_t)nt4 * 64) * MID_;
#pragma unroll
    for (int i = 0; i < 16; ++i) {
      int id = i * 128 + tid;
      g2l_async_b128(fds + id * 16, fsrc + id * 8);
    }
  }
  if (wv == 0) {
    // G chunk: contiguous 16 KB (1-D tile).  H chunk: 256 rows x 32 elems,
    // row stride HW_ (2-D tile).
    tdm_load_2d(sb0,         gbase, 8192u, 0u,   8192u);
    tdm_load_2d(sb0 + 16384, hbase, 32u,   256u, (uint64_t)HW_);
  }
  asm volatile("s_wait_asynccnt 0x0" ::: "memory");   // F landed (this wave)

  const f32x8 z8 = {0.f,0.f,0.f,0.f,0.f,0.f,0.f,0.f};
  f32x8 U[16];
#pragma unroll
  for (int ct = 0; ct < 16; ++ct) U[ct] = z8;
  float rowM[8], rowL[8];
#pragma unroll
  for (int r = 0; r < 8; ++r) { rowM[r] = -1e30f; rowL[r] = 0.f; }

  const __bf16* fl = Flds + (wv * 16 + l16) * 256 + half * 16; // A-frag base
  __bf16* pb = &Pb[wv][0];

  for (int j = 0; j < MCH_; ++j) {
    const int cur = j & 1;
    if (wv == 0) {    // issue next chunk's TDM, then wait current (in-order)
      if (j + 1 < MCH_) {
        uint32_t nb = sb0 + (1 - cur) * 32768;
        int m1 = (j + 1) * 32;
        tdm_load_2d(nb,         gbase + (size_t)m1 * MID_, 8192u, 0u, 8192u);
        tdm_load_2d(nb + 16384, hbase + m1,                32u, 256u,
                    (uint64_t)HW_);
        __builtin_amdgcn_s_wait_tensorcnt((short)2);
      } else {
        __builtin_amdgcn_s_wait_tensorcnt((short)0);
      }
    }
    __syncthreads();

    const __bf16* Gp = (const __bf16*)(&Sbuf[cur][0]);
    const __bf16* Hp = (const __bf16*)(&Sbuf[cur][16384]);

    // ---- S = (F^T G): two 16x16 tiles, K = 256 ----
    f32x8 S0 = z8, S1 = z8;
    const __bf16* gl_ = Gp + l16 * 256 + half * 16;
#pragma unroll
    for (int kc = 0; kc < 8; ++kc) {
      bf16x8 lo = *(const bf16x8*)(fl + kc * 32);
      bf16x8 hi = *(const bf16x8*)(fl + kc * 32 + 8);
      bf16x16 a  = CAT16(lo, hi);
      bf16x16 b0 = *(const bf16x16*)(gl_ + kc * 32);
      bf16x16 b1 = *(const bf16x16*)(gl_ + 16 * 256 + kc * 32);
      S0 = wmma_bf16(a, b0, S0);
      S1 = wmma_bf16(a, b1, S1);
    }

    // ---- online softmax over the 32 new columns ----
    float al[8];
#pragma unroll
    for (int r = 0; r < 8; ++r) {
      float s0 = S0[r] * SCALE_, s1 = S1[r] * SCALE_;
      float rm   = rowmax16(fmaxf(s0, s1));
      float mnew = fmaxf(rowM[r], rm);
      al[r] = __expf(rowM[r] - mnew);
      rowM[r] = mnew;
      s0 = __expf(s0 - mnew); s1 = __expf(s1 - mnew);
      S0[r] = s0; S1[r] = s1;
      rowL[r] = rowL[r] * al[r] + rowsum16(s0 + s1);
    }
#pragma unroll
    for (int ct = 0; ct < 16; ++ct)
#pragma unroll
      for (int r = 0; r < 8; ++r) U[ct][r] *= al[r];

    // ---- P (D-layout) -> bf16 A-frag via private LDS tile ----
#pragma unroll
    for (int r = 0; r < 8; ++r) {
      int row = r + 8 * half;
      pb[row * 32 + l16]      = (__bf16)S0[r];
      pb[row * 32 + 16 + l16] = (__bf16)S1[r];
    }
    asm volatile("s_wait_dscnt 0x0" ::: "memory"); // same-wave LDS RAW
    bf16x8 plo = *(const bf16x8*)(pb + l16 * 32 + half * 16);
    bf16x8 phi = *(const bf16x8*)(pb + l16 * 32 + 8 + half * 16);
    bf16x16 pA = CAT16(plo, phi);

    // ---- U += P * H^T  (16 c-tiles, K = 32) ----
    const __bf16* hl = Hp + l16 * 32 + half * 16;
#pragma unroll
    for (int ct = 0; ct < 16; ++ct) {
      bf16x16 bh = *(const bf16x16*)(hl + ct * 512);
      U[ct] = wmma_bf16(pA, bh, U[ct]);
    }
    __syncthreads();   // protect buffer cur before next overwrite
  }

  // ---- epilogue: normalize, transpose through LDS, coalesced z store ----
  float* zs = (float*)(&Sbuf[0][0]) + wv * 4096;   // 16 KB per wave
  float invL[8];
#pragma unroll
  for (int r = 0; r < 8; ++r) invL[r] = 1.f / rowL[r];
#pragma unroll
  for (int ct = 0; ct < 16; ++ct)
#pragma unroll
    for (int r = 0; r < 8; ++r)
      zs[(r + 8 * half) * 256 + ct * 16 + l16] = U[ct][r] * invL[r];
  asm volatile("s_wait_dscnt 0x0" ::: "memory");   // same-wave DS ordering
  {
    int row = lane >> 1, c0 = (lane & 1) * 128;
    const float* zr = zs + row * 256 + c0;
    const int nt = nt4 * 4 + wv;
    __bf16* zo = zT + ((size_t)b * HW_ + (size_t)nt * 16 + row) * MID_ + c0;
#pragma unroll
    for (int i = 0; i < 128; i += 8) {
      bf16x8 p;
#pragma unroll
      for (int k = 0; k < 8; ++k) p[k] = (__bf16)zr[i + k];
      *(bf16x8*)(zo + i) = p;
    }
  }
}

// =====================================================================
// Kernel 3: out = V z + v_b + x.  Grid (144, 16), 128 thr (4 waves),
// 8 oc-tiles per wave; z A-frags shared across oc-tiles.
// =====================================================================
__global__ __launch_bounds__(128)
void outproj_kernel(const __bf16* __restrict__ zT, const __bf16* __restrict__ wfV,
                    const float* __restrict__ v_b, const float* __restrict__ x,
                    float* __restrict__ out) {
  const int nt = blockIdx.x, b = blockIdx.y;
  const int lane = threadIdx.x & 31, wv = threadIdx.x >> 5;
  const int half = lane >> 4, l16 = lane & 15;

  bf16x16 az[8];
  const __bf16* zp = zT + ((size_t)b * HW_ + (size_t)nt * 16 + l16) * MID_ + half * 16;
#pragma unroll
  for (int kc = 0; kc < 8; ++kc) {
    bf16x8 lo = *(const bf16x8*)(zp + kc * 32);
    bf16x8 hi = *(const bf16x8*)(zp + kc * 32 + 8);
    az[kc] = CAT16(lo, hi);
  }

  const f32x8 z8 = {0.f,0.f,0.f,0.f,0.f,0.f,0.f,0.f};
#pragma unroll
  for (int t = 0; t < 8; ++t) {
    int ot = wv * 8 + t;
    f32x8 acc = z8;
#pragma unroll
    for (int kc = 0; kc < 8; ++kc) {
      const bf16x16* bv =
          (const bf16x16*)(wfV + (((size_t)ot * 8 + kc) * 32 + lane) * 16);
      acc = wmma_bf16(az[kc], *bv, acc);
    }
    int oc = ot * 16 + l16;
    size_t xo = ((size_t)b * C_ + oc) * HW_ + (size_t)nt * 16 + 8 * half;
    float bias = v_b[oc];
    f32x4 x0 = *(const f32x4*)(x + xo), x1 = *(const f32x4*)(x + xo + 4);
    f32x4 o0, o1;
#pragma unroll
    for (int r = 0; r < 4; ++r) {
      o0[r] = acc[r]     + bias + x0[r];
      o1[r] = acc[r + 4] + bias + x1[r];
    }
    *(f32x4*)(out + xo)     = o0;
    *(f32x4*)(out + xo + 4) = o1;
  }
}

// =====================================================================
// Host launcher
// =====================================================================
extern "C" void kernel_launch(void* const* d_in, const int* in_sizes, int n_in,
                              void* d_out, int out_size, void* d_ws, size_t ws_size,
                              hipStream_t stream) {
  const float* x       = (const float*)d_in[0];
  const float* f_w     = (const float*)d_in[1];
  const float* f_b     = (const float*)d_in[2];
  const float* f_gamma = (const float*)d_in[3];
  const float* f_beta  = (const float*)d_in[4];
  const float* f_mean  = (const float*)d_in[5];
  const float* f_var   = (const float*)d_in[6];
  const float* g_w     = (const float*)d_in[7];
  const float* g_b     = (const float*)d_in[8];
  const float* g_gamma = (const float*)d_in[9];
  const float* g_beta  = (const float*)d_in[10];
  const float* g_mean  = (const float*)d_in[11];
  const float* g_var   = (const float*)d_in[12];
  const float* h_w     = (const float*)d_in[13];
  const float* h_b     = (const float*)d_in[14];
  const float* v_w     = (const float*)d_in[15];
  const float* v_b     = (const float*)d_in[16];
  float* out = (float*)d_out;

  char* w = (char*)d_ws;
  const size_t FRAG = 262144;                 // 16*16*32*16 * 2B
  const size_t MAT  = (size_t)B_ * HW_ * MID_ * 2;  // 18,874,368 B
  __bf16* wfF  = (__bf16*)(w + 0 * FRAG);
  __bf16* wfG  = (__bf16*)(w + 1 * FRAG);
  __bf16* wfH  = (__bf16*)(w + 2 * FRAG);
  __bf16* wfV  = (__bf16*)(w + 3 * FRAG);
  float*  biasF = (float*)(w + 4 * FRAG);
  float*  biasG = (float*)(w + 4 * FRAG + 1024);
  char*   big  = w + 4 * FRAG + 2048;
  __bf16* fT = (__bf16*)(big + 0 * MAT);
  __bf16* gT = (__bf16*)(big + 1 * MAT);
  __bf16* hM = (__bf16*)(big + 2 * MAT);
  __bf16* zT = (__bf16*)(big + 3 * MAT);

  prep_kernel<<<2050, 256, 0, stream>>>(f_w, f_b, f_gamma, f_beta, f_mean, f_var,
                                        g_w, g_b, g_gamma, g_beta, g_mean, g_var,
                                        h_w, v_w, wfF, wfG, wfH, wfV, biasF, biasG);
  dim3 gridP(NT_, B_);
  proj_kernel<<<gridP, 128, 0, stream>>>(x, wfF, wfG, wfH, biasF, biasG, h_b,
                                         fT, gT, hM);
  dim3 gridA(NT_ / 4, B_);
  attn_kernel<<<gridA, 128, 0, stream>>>(fT, gT, hM, zT);
  outproj_kernel<<<gridP, 128, 0, stream>>>(zT, wfV, v_b, x, out);
}
